// AuctionRouter_52166672777639
// MI455X (gfx1250) — matrix operand order repaired and checked by reading, not compiled, not run on hardware
//
#include <hip/hip_runtime.h>
#include <hip/hip_bf16.h>
#include <math.h>

// MoE router: logits = x @ W^T + b -> softmax(64) -> top-2.
//
// Roofline (MI455X): x = 16384x4096 f32 = 268 MB streamed once from HBM
// @ 23.3 TB/s ~= 11.5 us floor; arithmetic intensity ~32 FLOP/B would need
// ~745 TFLOP/s in f32, so convert to f16 on the fly and use
// v_wmma_f32_16x16x32_f16 (f32 accumulate) -> compute time ~3 us, firmly
// memory-bound. W (1 MB) stays resident in the 192 MB L2; it is staged once
// per workgroup per K-step into LDS in WMMA B-operand order (double
// buffered, 1 barrier/iteration) so waves don't generate redundant L2 reads.
//
// TILE_M=64 (4 wave32 per WG, 256 WGs, ~29 KB LDS) so each WGP can host
// several workgroups -> more independent waves per SIMD to hide HBM latency.

typedef __attribute__((ext_vector_type(16))) _Float16 v16h;
typedef __attribute__((ext_vector_type(8)))  float    v8f;

#define TOKENS   16384
#define D_MODEL  4096
#define N_EXP    64
#define TILE_M   64           // tokens per workgroup (16 per wave)
#define NWAVES   4
#define NTHREADS 128
#define KSTEP    32           // K per wmma_f32_16x16x32_f16
#define NITER    (D_MODEL / KSTEP)
#define BSTRIDE  12           // dwords per (nb,lane) row in ldsB: 8 data + 4 pad (16B aligned)
#define BBUF     (4 * 32 * BSTRIDE)
#define LROW     65           // padded logits row stride (dwords) -> conflict-free column scans

__device__ __forceinline__ uint32_t pack_f16x2(float lo, float hi) {
  union { _Float16 h[2]; uint32_t u; } t;
  t.h[0] = (_Float16)lo;
  t.h[1] = (_Float16)hi;
  return t.u;
}

union BOpnd { uint4 q[2]; v16h h; };

__global__ __launch_bounds__(NTHREADS)
void router_topk_kernel(const float* __restrict__ x,
                        const float* __restrict__ W,
                        const float* __restrict__ bias,
                        float* __restrict__ out)
{
  // Double-buffered B tiles in operand order:
  // [buf (2)][expert_block nb (4)][b_lane (32)][8 packed f16 pairs + 4 pad]
  __shared__ uint32_t ldsB[2 * BBUF];                // 12 KB
  __shared__ float    ldsL[TILE_M * LROW];           // 16.6 KB logits (+bias)

  const int tid  = threadIdx.x;
  const int wave = tid >> 5;
  const int lane = tid & 31;
  const int half = lane >> 4;      // lane half selects K sub-window (K 16h..16h+15, same for A and B)
  const int lmod = lane & 15;
  const int wgBase = blockIdx.x * TILE_M;

  // A: this lane supplies token row (wave tile row = lane%16)
  const float* xrow = x + (size_t)(wgBase + wave * 16 + lmod) * D_MODEL;

  // --- B staging role: thread t handles expert e = t/2, K-half h = t&1
  //     (pairs p = 8h..8h+7, i.e. 16 consecutive K values). ---
  const int stE  = tid >> 1;           // expert 0..63
  const int stNb = stE >> 4;           // expert block
  const int stN  = stE & 15;           // column within block
  const int stH  = tid & 1;            // dest lane half (K 0-15 -> lanes 0-15, 16-31 -> 16-31)
  const float* wrow = W + (size_t)stE * D_MODEL + 16 * stH;
  uint32_t* stDst = &ldsB[(stNb * 32 + stH * 16 + stN) * BSTRIDE];

  v8f acc0 = {}, acc1 = {}, acc2 = {}, acc3 = {};

  // Stage first K-step into buffer 0.
  {
    float4 w0 = *(const float4*)(wrow + 0);
    float4 w1 = *(const float4*)(wrow + 4);
    float4 w2 = *(const float4*)(wrow + 8);
    float4 w3 = *(const float4*)(wrow + 12);
    uint4 lo, hi;
    lo.x = pack_f16x2(w0.x, w0.y); lo.y = pack_f16x2(w0.z, w0.w);
    lo.z = pack_f16x2(w1.x, w1.y); lo.w = pack_f16x2(w1.z, w1.w);
    hi.x = pack_f16x2(w2.x, w2.y); hi.y = pack_f16x2(w2.z, w2.w);
    hi.z = pack_f16x2(w3.x, w3.y); hi.w = pack_f16x2(w3.z, w3.w);
    *(uint4*)stDst = lo;
    *(uint4*)(stDst + 4) = hi;
  }

  for (int it = 0; it < NITER; ++it) {
    const int k0 = it * KSTEP;
    __syncthreads();   // publishes staging of buf[it&1]; protects buf[(it+1)&1] reuse

    // Stage next K-step into the other buffer (overlaps with compute below).
    if (it + 1 < NITER) {
      const int kn = k0 + KSTEP;
      uint32_t* dst = stDst + ((it + 1) & 1) * BBUF;
      float4 w0 = *(const float4*)(wrow + kn);
      float4 w1 = *(const float4*)(wrow + kn + 4);
      float4 w2 = *(const float4*)(wrow + kn + 8);
      float4 w3 = *(const float4*)(wrow + kn + 12);
      uint4 lo, hi;
      lo.x = pack_f16x2(w0.x, w0.y); lo.y = pack_f16x2(w0.z, w0.w);
      lo.z = pack_f16x2(w1.x, w1.y); lo.w = pack_f16x2(w1.z, w1.w);
      hi.x = pack_f16x2(w2.x, w2.y); hi.y = pack_f16x2(w2.z, w2.w);
      hi.z = pack_f16x2(w3.x, w3.y); hi.w = pack_f16x2(w3.z, w3.w);
      *(uint4*)dst = lo;
      *(uint4*)(dst + 4) = hi;
    }

    // Prefetch upcoming x data (gfx1250 global_prefetch path).
    __builtin_prefetch(xrow + k0 + KSTEP * 4, 0, 0);

    // --- A operand: contiguous-K 16x32 layout (same K order as B) ---
    // lanes 0-15 : v0-7 = K 0..15
    // lanes 16-31: v0-7 = K 16..31
    float4 a0 = *(const float4*)(xrow + k0 + half * 16);
    float4 a1 = *(const float4*)(xrow + k0 + half * 16 + 4);
    float4 a2 = *(const float4*)(xrow + k0 + half * 16 + 8);
    float4 a3 = *(const float4*)(xrow + k0 + half * 16 + 12);
    v16h A;
    A[0]  = (_Float16)a0.x; A[1]  = (_Float16)a0.y;
    A[2]  = (_Float16)a0.z; A[3]  = (_Float16)a0.w;
    A[4]  = (_Float16)a1.x; A[5]  = (_Float16)a1.y;
    A[6]  = (_Float16)a1.z; A[7]  = (_Float16)a1.w;
    A[8]  = (_Float16)a2.x; A[9]  = (_Float16)a2.y;
    A[10] = (_Float16)a2.z; A[11] = (_Float16)a2.w;
    A[12] = (_Float16)a3.x; A[13] = (_Float16)a3.y;
    A[14] = (_Float16)a3.z; A[15] = (_Float16)a3.w;

    // --- B operands from LDS (already f16, operand order) ---
    const uint32_t* bbase = &ldsB[(it & 1) * BBUF + lane * BSTRIDE];
    BOpnd b0u, b1u, b2u, b3u;
    b0u.q[0] = *(const uint4*)(bbase + 0 * 32 * BSTRIDE);
    b0u.q[1] = *(const uint4*)(bbase + 0 * 32 * BSTRIDE + 4);
    b1u.q[0] = *(const uint4*)(bbase + 1 * 32 * BSTRIDE);
    b1u.q[1] = *(const uint4*)(bbase + 1 * 32 * BSTRIDE + 4);
    b2u.q[0] = *(const uint4*)(bbase + 2 * 32 * BSTRIDE);
    b2u.q[1] = *(const uint4*)(bbase + 2 * 32 * BSTRIDE + 4);
    b3u.q[0] = *(const uint4*)(bbase + 3 * 32 * BSTRIDE);
    b3u.q[1] = *(const uint4*)(bbase + 3 * 32 * BSTRIDE + 4);

    acc0 = __builtin_amdgcn_wmma_f32_16x16x32_f16(false, A, false, b0u.h, (short)0, acc0, false, false);
    acc1 = __builtin_amdgcn_wmma_f32_16x16x32_f16(false, A, false, b1u.h, (short)0, acc1, false, false);
    acc2 = __builtin_amdgcn_wmma_f32_16x16x32_f16(false, A, false, b2u.h, (short)0, acc2, false, false);
    acc3 = __builtin_amdgcn_wmma_f32_16x16x32_f16(false, A, false, b3u.h, (short)0, acc3, false, false);
  }

  // --- Spill logits (+bias) to LDS. C/D layout: vgpr r -> M=r (lanes 0-15) / M=8+r (16-31),
  //     N = lane % 16. ---
  const float bs0 = bias[lmod];
  const float bs1 = bias[16 + lmod];
  const float bs2 = bias[32 + lmod];
  const float bs3 = bias[48 + lmod];
  const int tokBase = wave * 16 + half * 8;
#pragma unroll
  for (int r = 0; r < 8; ++r) {
    float* lr = &ldsL[(tokBase + r) * LROW + lmod];
    lr[0]  = acc0[r] + bs0;
    lr[16] = acc1[r] + bs1;
    lr[32] = acc2[r] + bs2;
    lr[48] = acc3[r] + bs3;
  }
  __syncthreads();

  // --- Per-token softmax + top-2 (strict '>' scan matches jax top_k tie order) ---
  if (tid < TILE_M) {
    const float* lr = &ldsL[tid * LROW];
    float m1 = -INFINITY, m2 = -INFINITY;
    int i1 = 0, i2 = 0;
    for (int e = 0; e < N_EXP; ++e) {
      float v = lr[e];
      if (v > m1)      { m2 = m1; i2 = i1; m1 = v; i1 = e; }
      else if (v > m2) { m2 = v; i2 = e; }
    }
    float s = 0.0f;
    for (int e = 0; e < N_EXP; ++e) s += __expf(lr[e] - m1);
    const float inv = 1.0f / s;

    const int gt = wgBase + tid;
    // Output 0: top-k indices (written as float; tuple flattening promotes to f32)
    out[2 * gt + 0] = (float)i1;
    out[2 * gt + 1] = (float)i2;
    // Output 1: top-k softmax scores
    out[2 * TOKENS + 2 * gt + 0] = inv;                      // exp(m1-m1)/s
    out[2 * TOKENS + 2 * gt + 1] = __expf(m2 - m1) * inv;
  }
}

extern "C" void kernel_launch(void* const* d_in, const int* in_sizes, int n_in,
                              void* d_out, int out_size, void* d_ws, size_t ws_size,
                              hipStream_t stream) {
  (void)in_sizes; (void)n_in; (void)out_size; (void)d_ws; (void)ws_size;
  const float* x = (const float*)d_in[0];
  const float* W = (const float*)d_in[1];
  const float* b = (const float*)d_in[2];
  float* out = (float*)d_out;

  dim3 grid(TOKENS / TILE_M);   // 256 workgroups
  dim3 block(NTHREADS);         // 4 wave32
  hipLaunchKernelGGL(router_topk_kernel, grid, block, 0, stream, x, W, b, out);
}
